// MLPBlock_45071386804514
// MI455X (gfx1250) — matrix operand order repaired and checked by reading, not compiled, not run on hardware
//
#include <hip/hip_runtime.h>
#include <hip/hip_bf16.h>

#define N_NODES 100000
#define N_EDGES 3200000
#define D       64
#define K3      192      // 3*D
#define HID     36
#define HIDP    48       // HID padded to 3 WMMA column tiles
#define OUT_C   2
#define NTILES  (N_NODES / 16)   // 6250, exact
#define WAVES   4                // waves per block in MLP kernel

typedef __attribute__((ext_vector_type(16))) __bf16 v16bf;
typedef __attribute__((ext_vector_type(8)))  float  v8f;

union Frag {
    v16bf bf;
    unsigned int u[8];
};

__device__ __forceinline__ unsigned short f2bf(float x) {
    unsigned int u = __float_as_uint(x);
    unsigned int r = u + 0x7FFFu + ((u >> 16) & 1u);   // round-to-nearest-even
    return (unsigned short)(r >> 16);
}

// ---------------- kernel 1: zero the sent/recv accumulators ----------------
__global__ void zero_kernel(float4* __restrict__ ws, int n4) {
    int i = blockIdx.x * blockDim.x + threadIdx.x;
    if (i < n4) ws[i] = make_float4(0.f, 0.f, 0.f, 0.f);
}

// ---------------- kernel 2: dual segment-sum via L2 atomics ----------------
// one wave per edge; lane p owns elements 2p, 2p+1 of the 64-wide edge row.
// edges are streamed with non-temporal loads so the 819MB stream does not
// evict the 51MB sent/recv accumulators out of the 192MB L2.
__global__ void scatter_kernel(const float* __restrict__ edges,
                               const int* __restrict__ senders,
                               const int* __restrict__ receivers,
                               float* __restrict__ sent,
                               float* __restrict__ recv) {
    unsigned int gtid = blockIdx.x * blockDim.x + threadIdx.x;
    unsigned int e = gtid >> 5;
    unsigned int p = gtid & 31u;
    if (e >= N_EDGES) return;

    const double* ep = (const double*)(edges + (size_t)e * D + 2u * p);
    double dv = __builtin_nontemporal_load(ep);          // TH=NT 64-bit load
    float2 v = *(float2*)&dv;

    int s = senders[e];
    int r = receivers[e];
    size_t so = (size_t)s * D + 2u * p;
    size_t ro = (size_t)r * D + 2u * p;
    atomicAdd(&sent[so],     v.x);
    atomicAdd(&sent[so + 1], v.y);
    atomicAdd(&recv[ro],     v.x);
    atomicAdd(&recv[ro + 1], v.y);
}

// ---------------- kernel 3: fused concat + MLP via WMMA bf16 ----------------
__global__ __launch_bounds__(WAVES * 32)
void mlp_kernel(const float* __restrict__ nodes,
                const float* __restrict__ sent,
                const float* __restrict__ recv,
                const float* __restrict__ W1,
                const float* __restrict__ b1,
                const float* __restrict__ W2,
                const float* __restrict__ b2,
                float* __restrict__ out) {
    // bf16 feature tiles: [wave][16 rows][192 k], row-major (k contiguous)
    __shared__ unsigned short featsLds[WAVES][16 * K3];      // 24576 B
    // W1 transposed + padded: [48 n][192 k], k contiguous
    __shared__ unsigned short w1tLds[HIDP * K3];             // 18432 B
    // hidden activations: [wave][16 rows][36]
    __shared__ float hLds[WAVES][16 * HID];                  //  9216 B

    const int tid  = threadIdx.x;
    const int wave = tid >> 5;
    const int lane = tid & 31;
    const int tile = blockIdx.x * WAVES + wave;
    const bool active = (tile < NTILES);
    const int row0 = tile * 16;

    // ---- stage W1^T as bf16 (whole block, once) ----
    for (int idx = tid; idx < HIDP * K3; idx += blockDim.x) {
        int n = idx / K3;
        int k = idx - n * K3;
        float w = (n < HID) ? W1[k * HID + n] : 0.0f;
        w1tLds[idx] = f2bf(w);
    }

    // ---- stage this wave's 16x192 feature tile as bf16 ----
    for (int f = lane; f < 16 * K3; f += 32) {
        int m = f / K3;
        int k = f - m * K3;
        float val = 0.0f;
        if (active) {
            int node = row0 + m;
            val = (k < D)     ? nodes[(size_t)node * D + k]
                : (k < 2 * D) ? sent [(size_t)node * D + (k - D)]
                              : recv [(size_t)node * D + (k - 2 * D)];
        }
        featsLds[wave][f] = f2bf(val);
    }
    __syncthreads();

    const int m    = lane & 15;   // N column (B/C) or M row (A)
    const int half = lane >> 4;

    // ---- build 6 A fragments (16x32 bf16 each), per documented wave32 layout:
    // lanes 0-15: VGPR0-3 -> K 0..7, VGPR4-7 -> K 16..23 (half=0)
    // lanes16-31: VGPR0-3 -> K 8..15, VGPR4-7 -> K 24..31 (half=1)
    Frag a[6];
    {
        const unsigned short* fb = &featsLds[wave][m * K3];
        #pragma unroll
        for (int kk = 0; kk < 6; ++kk) {
            int kb = kk * 32;
            #pragma unroll
            for (int i = 0; i < 8; ++i) {
                int k = kb + ((i < 4) ? 0 : 16) + 8 * half + 2 * (i & 3);
                a[kk].u[i] = *(const unsigned int*)(fb + k);   // ds_load_b32
            }
        }
    }

    // ---- 3 column tiles of 16, K = 192 in 6 WMMA steps each ----
    #pragma unroll
    for (int ct = 0; ct < 3; ++ct) {
        const int ncol = ct * 16 + m;
        const unsigned short* wb = &w1tLds[ncol * K3];

        v8f acc = {};
        #pragma unroll
        for (int kk = 0; kk < 6; ++kk) {
            // B 32x16 bf16 layout (same K order as A): lanes0-15 VGPR0-3 K 0..7, VGPR4-7 K 16..23;
            // lanes16-31 VGPR0-3 K 8..15, VGPR4-7 K 24..31
            Frag bfrag;
            #pragma unroll
            for (int i = 0; i < 8; ++i) {
                int k = kk * 32 + ((i < 4) ? 0 : 16) + 8 * half + 2 * (i & 3);
                bfrag.u[i] = *(const unsigned int*)(wb + k);   // ds_load_b32
            }
            acc = __builtin_amdgcn_wmma_f32_16x16x32_bf16(
                /*neg_a=*/false, a[kk].bf,
                /*neg_b=*/false, bfrag.bf,
                /*c_mod=*/(short)0, acc,
                /*reuse_a=*/false, /*reuse_b=*/false);
        }

        // bias + ReLU; C layout: VGPR r -> M = r + 8*half, N = lane&15
        float bias = (ncol < HID) ? b1[ncol] : 0.0f;
        #pragma unroll
        for (int r = 0; r < 8; ++r) {
            float hv = fmaxf(acc[r] + bias, 0.0f);
            int mm = r + 8 * half;
            if (ncol < HID) hLds[wave][mm * HID + ncol] = hv;
        }
    }
    __syncthreads();

    // ---- second layer: 16 rows x 2 outputs = 32 results, one per lane ----
    if (active) {
        int mrow = lane & 15;
        int o    = lane >> 4;
        float accf = b2[o];
        #pragma unroll
        for (int n = 0; n < HID; ++n)
            accf += hLds[wave][mrow * HID + n] * W2[n * OUT_C + o];
        out[(size_t)(row0 + mrow) * OUT_C + o] = accf;
    }
}

extern "C" void kernel_launch(void* const* d_in, const int* in_sizes, int n_in,
                              void* d_out, int out_size, void* d_ws, size_t ws_size,
                              hipStream_t stream) {
    const float* nodes     = (const float*)d_in[0];
    const float* edges     = (const float*)d_in[1];
    const int*   senders   = (const int*)  d_in[2];
    const int*   receivers = (const int*)  d_in[3];
    const float* W1        = (const float*)d_in[4];
    const float* b1        = (const float*)d_in[5];
    const float* W2        = (const float*)d_in[6];
    const float* b2        = (const float*)d_in[7];
    float* out = (float*)d_out;

    float* sent = (float*)d_ws;
    float* recv = sent + (size_t)N_NODES * D;

    // 1) zero accumulators (workspace is poisoned; must re-init every call)
    {
        int n4 = (2 * N_NODES * D) / 4;               // 3,200,000 float4
        int blocks = (n4 + 255) / 256;
        zero_kernel<<<blocks, 256, 0, stream>>>((float4*)d_ws, n4);
    }
    // 2) dual segment-sum
    {
        long long threads = (long long)N_EDGES * 32;  // 102,400,000
        int blocks = (int)((threads + 255) / 256);    // 400,000
        scatter_kernel<<<blocks, 256, 0, stream>>>(edges, senders, receivers,
                                                   sent, recv);
    }
    // 3) fused concat + MLP (WMMA)
    {
        int blocks = (NTILES + WAVES - 1) / WAVES;    // 1563
        mlp_kernel<<<blocks, WAVES * 32, 0, stream>>>(nodes, sent, recv,
                                                      W1, b1, W2, b2, out);
    }
}
